// RWKV_TimeMix_67095979099113
// MI455X (gfx1250) — compile-verified, matrix operand order checked by reading, not run on hardware
//
#include <hip/hip_runtime.h>
#include <hip/hip_bf16.h>

// ---------------------------------------------------------------------------
// RWKV TimeMix for MI455X (gfx1250, wave32, WMMA)
//   B=4, T=2048, C=1024
//   Projections:      v_wmma_f32_16x16x32_f16 (f16 in, f32 accumulate)
//   Depthwise conv:   v_wmma_f32_16x16x4_f32  (full fp32 Toeplitz matmul,
//                     causal mask via zero-padded w row in LDS)
//   Conv staging:     global_load_async_to_lds_b128 (ASYNCcnt path)
// ---------------------------------------------------------------------------

#define USE_ASYNC_LDS 1

typedef _Float16 v16h __attribute__((ext_vector_type(16)));
typedef float    v8f  __attribute__((ext_vector_type(8)));
typedef float    v2f  __attribute__((ext_vector_type(2)));

#define RW_B 4
#define RW_T 2048
#define RW_C 1024
#define RW_M (RW_B * RW_T)      // 8192
#define K_MAX_F 30.0f
#define B_EPS_F 1e-8f
#define SIG_LD 2056             // padded row stride (breaks 2048%64==0 bank aliasing)

// Load a 16x32 f16 fragment (A-matrix layout, ISA 7.12.2):
//  lane<16 : row = lane,    K = {0..7} then {16..23}
//  lane>=16: row = lane-16, K = {8..15} then {24..31}
__device__ __forceinline__ v16h load_frag(const _Float16* p) {
    union { v16h h; float4 f4[2]; } u;
    u.f4[0] = *reinterpret_cast<const float4*>(p);        // K +0..7  (16B)
    u.f4[1] = *reinterpret_cast<const float4*>(p + 16);   // K +16..23 (16B)
    return u.h;
}

// Async global -> LDS copy of 16 bytes (tracked by ASYNCcnt).
__device__ __forceinline__ void async_copy16(const float* g, float* lds_ptr) {
#if USE_ASYNC_LDS
    unsigned lo = (unsigned)(size_t)lds_ptr;   // low 32 bits = LDS offset
    asm volatile("global_load_async_to_lds_b128 %0, %1, off"
                 :: "v"(lo), "v"(g) : "memory");
#else
    *(float4*)lds_ptr = *(const float4*)g;
#endif
}

__device__ __forceinline__ void async_wait_all() {
#if USE_ASYNC_LDS
    asm volatile("s_wait_asynccnt 0" ::: "memory");
#endif
}

// ---------------------------------------------------------------------------
// 1) time-shift mixing: xk/xv/xr (f16) from x (f32)
// ---------------------------------------------------------------------------
__global__ __launch_bounds__(256) void mix_kernel(
    const float* __restrict__ x,
    const float* __restrict__ tmk, const float* __restrict__ tmv,
    const float* __restrict__ tmr,
    _Float16* __restrict__ xk, _Float16* __restrict__ xv,
    _Float16* __restrict__ xr)
{
    long i = (long)blockIdx.x * 256 + threadIdx.x;        // over B*T*C
    int  c = (int)(i % RW_C);
    long bt = i / RW_C;
    int  t = (int)(bt % RW_T);
    float xi = x[i];
    float xp = (t > 0) ? x[i - RW_C] : 0.0f;
    float mk = tmk[c], mv = tmv[c], mr = tmr[c];
    xk[i] = (_Float16)(xi * mk + xp * (1.0f - mk));
    xv[i] = (_Float16)(xi * mv + xp * (1.0f - mv));
    xr[i] = (_Float16)(xi * mr + xp * (1.0f - mr));
}

// ---------------------------------------------------------------------------
// 2) f32 -> f16 weight conversion
// ---------------------------------------------------------------------------
__global__ __launch_bounds__(256) void cvt_kernel(
    const float* __restrict__ src, _Float16* __restrict__ dst)
{
    long i = (long)blockIdx.x * 256 + threadIdx.x;
    dst[i] = (_Float16)src[i];
}

// ---------------------------------------------------------------------------
// 3) GEMM  out[m,n] = sum_k A[m,k] * W[n,k]   (M=8192, N=K=1024)
//    mode 0: out[m][n]  (row-major, f32)
//    mode 1: out[b][n][t] with m = b*T + t  (transposed per batch, f32)
// ---------------------------------------------------------------------------
__global__ __launch_bounds__(256) void gemm_f16_kernel(
    const _Float16* __restrict__ A, const _Float16* __restrict__ W,
    float* __restrict__ out, int M, int N, int K, int mode)
{
    const int lane = threadIdx.x & 31;
    const int wave = threadIdx.x >> 5;
    const int m_base = blockIdx.x * 128 + (wave >> 1) * 32;
    const int n_base = blockIdx.y * 64 + (wave & 1) * 32;
    const int r15  = lane & 15;
    const int koff = (lane >> 4) * 8;

    v8f acc00 = {}; v8f acc01 = {}; v8f acc10 = {}; v8f acc11 = {};

    const _Float16* a0p = A + (size_t)(m_base      + r15) * K + koff;
    const _Float16* a1p = A + (size_t)(m_base + 16 + r15) * K + koff;
    const _Float16* b0p = W + (size_t)(n_base      + r15) * K + koff;
    const _Float16* b1p = W + (size_t)(n_base + 16 + r15) * K + koff;

    for (int k0 = 0; k0 < K; k0 += 32) {
        v16h a0 = load_frag(a0p + k0);
        v16h a1 = load_frag(a1p + k0);
        v16h b0 = load_frag(b0p + k0);
        v16h b1 = load_frag(b1p + k0);
        acc00 = __builtin_amdgcn_wmma_f32_16x16x32_f16(false, a0, false, b0,
                                                       (short)0, acc00, false, false);
        acc01 = __builtin_amdgcn_wmma_f32_16x16x32_f16(false, a0, false, b1,
                                                       (short)0, acc01, false, false);
        acc10 = __builtin_amdgcn_wmma_f32_16x16x32_f16(false, a1, false, b0,
                                                       (short)0, acc10, false, false);
        acc11 = __builtin_amdgcn_wmma_f32_16x16x32_f16(false, a1, false, b1,
                                                       (short)0, acc11, false, false);
    }

    // C/D layout: VGPR r -> M = r + (lane>=16 ? 8 : 0); N = lane & 15.
    const int mrow = (lane >> 4) * 8;
    for (int r = 0; r < 8; ++r) {
        int mm0 = m_base + mrow + r;
        int mm1 = m_base + 16 + mrow + r;
        int nn0 = n_base + r15;
        int nn1 = n_base + 16 + r15;
        if (mode == 0) {
            out[(size_t)mm0 * N + nn0] = acc00[r];
            out[(size_t)mm0 * N + nn1] = acc01[r];
            out[(size_t)mm1 * N + nn0] = acc10[r];
            out[(size_t)mm1 * N + nn1] = acc11[r];
        } else {
            int b0i = mm0 / RW_T, t0i = mm0 % RW_T;
            int b1i = mm1 / RW_T, t1i = mm1 % RW_T;
            out[((size_t)b0i * N + nn0) * RW_T + t0i] = acc00[r];
            out[((size_t)b0i * N + nn1) * RW_T + t0i] = acc01[r];
            out[((size_t)b1i * N + nn0) * RW_T + t1i] = acc10[r];
            out[((size_t)b1i * N + nn1) * RW_T + t1i] = acc11[r];
        }
    }
}

// ---------------------------------------------------------------------------
// 4) causal depthwise conv via f32 WMMA (Toeplitz matmul).
//    Staging: async global->LDS b128 of the w row and raw k/v rows,
//    then in-place LDS transform: sig[b]=exp(min(k,30))*v, sig[4+b]=exp(..).
//    D tile: M=16 times x N=16 (cols 0-3 wkv_b, 4-7 wk_b, 8-15 ignored),
//    K = lag s in chunks of 4:  A[t][k] = w[T-1-t+s0+k], B[k][n] = sig_n[s0+k]
// ---------------------------------------------------------------------------
__global__ __launch_bounds__(256) void conv_wmma_kernel(
    const float* __restrict__ wrow,   // time_w [C][1][T]
    const float* __restrict__ kbuf,   // raw k  [b][c][t]
    const float* __restrict__ vbuf,   // raw v  [b][c][t]
    const float* __restrict__ rbuf,   // r      [b][t][c]
    _Float16* __restrict__ rwkv)      // out    [b][t][c] (f16)
{
    __shared__ __align__(16) float w_s[RW_T + 32];
    __shared__ __align__(16) float sig[8][SIG_LD];

    const int c = blockIdx.x;

    // zero pad above T-1 => causal masking comes for free
    for (int i = RW_T + threadIdx.x; i < RW_T + 32; i += 256)
        w_s[i] = 0.0f;

    // async-stage: w row -> w_s, v rows -> sig[0..3], k rows -> sig[4..7]
    {
        const float* wsrc = wrow + (size_t)c * RW_T;
        for (int chunk = threadIdx.x; chunk < RW_T / 4; chunk += 256)
            async_copy16(wsrc + chunk * 4, &w_s[chunk * 4]);
        for (int b = 0; b < RW_B; ++b) {
            size_t base = ((size_t)b * RW_C + c) * RW_T;
            const float* vsrc = vbuf + base;
            const float* ksrc = kbuf + base;
            for (int chunk = threadIdx.x; chunk < RW_T / 4; chunk += 256) {
                async_copy16(vsrc + chunk * 4, &sig[b][chunk * 4]);
                async_copy16(ksrc + chunk * 4, &sig[4 + b][chunk * 4]);
            }
        }
    }
    async_wait_all();
    __syncthreads();

    // in-place transform: kexp = exp(min(k,30)); kv = kexp * v
    for (int b = 0; b < RW_B; ++b) {
        for (int t = threadIdx.x; t < RW_T; t += 256) {
            float ke = __expf(fminf(sig[4 + b][t], K_MAX_F));
            sig[4 + b][t] = ke;
            sig[b][t]     = ke * sig[b][t];
        }
    }
    __syncthreads();

    const int lane = threadIdx.x & 31;
    const int wave = threadIdx.x >> 5;
    const int row  = lane & 15;
    const int kg   = (lane >> 4) << 1;   // K offset {0,2} for this half-wave
    const int nsig = lane & 7;           // signal row for B frag (cols 8-15 dup, ignored)

    // 128 t-tiles of 16, strided across the 8 waves for load balance
    for (int tile = wave; tile < RW_T / 16; tile += 8) {
        const int t0 = tile * 16;
        const int wbase = (RW_T - 1) - (t0 + row) + kg;
        v8f acc = {};
        for (int s0 = 0; s0 < t0 + 16; s0 += 4) {
            v2f a, b;
            int wi = wbase + s0;
            a[0] = w_s[wi];
            a[1] = w_s[wi + 1];
            b[0] = sig[nsig][s0 + kg];
            b[1] = sig[nsig][s0 + kg + 1];
            acc = __builtin_amdgcn_wmma_f32_16x16x4_f32(false, a, false, b,
                                                        (short)0, acc, false, false);
        }
        // D layout: VGPR r -> M = r + (lane>=16 ? 8 : 0); N = lane & 15.
        // col n<4: wkv for batch n; col n+4: wk for batch n.
        for (int r = 0; r < 8; ++r) {
            float wkv = acc[r];
            int src = (lane & 16) | ((lane + 4) & 15);   // partner holding wk
            float wkp = __shfl(wkv, src, 32);
            if ((lane & 15) < 4) {
                int b  = lane & 3;
                int t  = t0 + r + ((lane >> 4) << 3);
                float wk = wkp + B_EPS_F;
                float rv = rbuf[((size_t)b * RW_T + t) * RW_C + c];
                float sg = 1.0f / (1.0f + __expf(-rv));
                rwkv[((size_t)b * RW_T + t) * RW_C + c] =
                    (_Float16)(sg * wkv / wk);
            }
        }
    }
}

// ---------------------------------------------------------------------------
// launch
// ---------------------------------------------------------------------------
extern "C" void kernel_launch(void* const* d_in, const int* in_sizes, int n_in,
                              void* d_out, int out_size, void* d_ws, size_t ws_size,
                              hipStream_t stream) {
    const float* x   = (const float*)d_in[0];
    const float* tmk = (const float*)d_in[1];
    const float* tmv = (const float*)d_in[2];
    const float* tmr = (const float*)d_in[3];
    const float* tw  = (const float*)d_in[4];
    // d_in[5] = time_ww (unused by the reference)
    const float* Wk  = (const float*)d_in[6];
    const float* Wv  = (const float*)d_in[7];
    const float* Wr  = (const float*)d_in[8];
    const float* Wo  = (const float*)d_in[9];
    float* out = (float*)d_out;

    const long BTC = (long)RW_B * RW_T * RW_C;            // 8,388,608
    const long CC  = (long)RW_C * RW_C;                   // 1,048,576

    char* ws = (char*)d_ws;
    _Float16* xk16 = (_Float16*)(ws);                     // BTC f16
    _Float16* xv16 = xk16 + BTC;
    _Float16* xr16 = xv16 + BTC;
    _Float16* w16  = xr16 + BTC;                          // 4*CC f16
    float*    kbuf = (float*)(w16 + 4 * CC);              // [b][c][t]
    float*    vbuf = kbuf + BTC;                          // [b][c][t]
    float*    rbuf = vbuf + BTC;                          // [b][t][c]
    _Float16* rwkv16 = xk16;                              // reuse (xk dead by then)

    // 1) mixing
    mix_kernel<<<(int)(BTC / 256), 256, 0, stream>>>(x, tmk, tmv, tmr,
                                                     xk16, xv16, xr16);
    // 2) weight conversion
    cvt_kernel<<<(int)(CC / 256), 256, 0, stream>>>(Wk, w16);
    cvt_kernel<<<(int)(CC / 256), 256, 0, stream>>>(Wv, w16 + CC);
    cvt_kernel<<<(int)(CC / 256), 256, 0, stream>>>(Wr, w16 + 2 * CC);
    cvt_kernel<<<(int)(CC / 256), 256, 0, stream>>>(Wo, w16 + 3 * CC);

    // 3) projections: k, v -> [b][c][t]; r -> [b][t][c]
    dim3 gg(RW_M / 128, RW_C / 64);
    gemm_f16_kernel<<<gg, 256, 0, stream>>>(xk16, w16,          kbuf,
                                            RW_M, RW_C, RW_C, 1);
    gemm_f16_kernel<<<gg, 256, 0, stream>>>(xv16, w16 + CC,     vbuf,
                                            RW_M, RW_C, RW_C, 1);
    gemm_f16_kernel<<<gg, 256, 0, stream>>>(xr16, w16 + 2 * CC, rbuf,
                                            RW_M, RW_C, RW_C, 0);

    // 4) causal depthwise conv (f32 WMMA, async-LDS staging) -> rwkv (f16)
    conv_wmma_kernel<<<RW_C, 256, 0, stream>>>(tw, kbuf, vbuf, rbuf, rwkv16);

    // 5) output projection
    gemm_f16_kernel<<<gg, 256, 0, stream>>>(rwkv16, w16 + 3 * CC, out,
                                            RW_M, RW_C, RW_C, 0);
}
